// RecurrentNet_87076166960190
// MI455X (gfx1250) — matrix operand order repaired and checked by reading, not compiled, not run on hardware
//
#include <hip/hip_runtime.h>
#include <hip/hip_bf16.h>
#include <math.h>

typedef __attribute__((ext_vector_type(16))) _Float16 v16h;
typedef __attribute__((ext_vector_type(8)))  _Float16 v8h;
typedef __attribute__((ext_vector_type(4)))  _Float16 v4h;
typedef __attribute__((ext_vector_type(8)))  float    v8f;

#define T_SEQ 512
#define BATCH 256
#define INDIM 248
#define HID   1024
#define KP    1280      // combined K: 1024 (h) + 248 (x) + 8 pad
#define ULD   1288      // LDS row stride in halves (bank-spread padding)
#define NBLK  64
#define NTHR  256
#define LDS_BYTES (64 * ULD * 2)   // 64-row activation slab, f16

#if __has_builtin(__builtin_amdgcn_tanhf)
#define TANH(x) __builtin_amdgcn_tanhf(x)
#else
#define TANH(x) tanhf(x)
#endif

// workspace byte offsets
#define WS_SYNC   0                          // 2 x u32 (256 B reserved)
#define WS_BIAS   256                        // 1024 f32
#define WS_Z      8192                       // 512*256 f32
#define WS_H0     (8192 + T_SEQ*BATCH*4)     // 256*1024 f16
#define WS_H1     (WS_H0 + BATCH*HID*2)      // 256*1024 f16
#define WS_WC     (WS_H1 + BATCH*HID*2)      // 1024*1280 f16

// ---------------------------------------------------------------- setup
__global__ void setup_wc(const float* __restrict__ W_hh,
                         const float* __restrict__ W_ih,
                         unsigned char* __restrict__ ws) {
  _Float16* Wc = (_Float16*)(ws + WS_WC);
  int i = blockIdx.x * blockDim.x + threadIdx.x;
  if (i >= HID * KP) return;
  int n = i / KP, k = i - n * KP;
  float v = 0.f;
  if (k < HID)              v = W_hh[n * HID + k];
  else if (k < HID + INDIM) v = W_ih[n * INDIM + (k - HID)];
  Wc[i] = (_Float16)v;
}

__global__ void setup_misc(const float* __restrict__ b_ih,
                           const float* __restrict__ b_hh,
                           unsigned char* __restrict__ ws) {
  int i = blockIdx.x * blockDim.x + threadIdx.x;
  unsigned* sync = (unsigned*)(ws + WS_SYNC);
  float* bias = (float*)(ws + WS_BIAS);
  float* z    = (float*)(ws + WS_Z);
  unsigned* h0 = (unsigned*)(ws + WS_H0);      // zero f16 h0 as u32 pairs
  if (i < 2) sync[i] = 0u;
  if (i < HID) bias[i] = b_ih[i] + b_hh[i];
  if (i < T_SEQ * BATCH) z[i] = 0.f;
  if (i < BATCH * HID / 2) h0[i] = 0u;
}

// ---------------------------------------------------------------- grid barrier
__device__ __forceinline__ void grid_sync(unsigned* cnt, unsigned* gen) {
  __threadfence();                 // release h stores to device scope
  __syncthreads();
  if (threadIdx.x == 0) {
    unsigned g = __hip_atomic_load(gen, __ATOMIC_RELAXED, __HIP_MEMORY_SCOPE_AGENT);
    unsigned a = __hip_atomic_fetch_add(cnt, 1u, __ATOMIC_ACQ_REL, __HIP_MEMORY_SCOPE_AGENT);
    if (a == NBLK - 1) {
      __hip_atomic_store(cnt, 0u, __ATOMIC_RELAXED, __HIP_MEMORY_SCOPE_AGENT);
      __hip_atomic_fetch_add(gen, 1u, __ATOMIC_ACQ_REL, __HIP_MEMORY_SCOPE_AGENT);
    } else {
      while (__hip_atomic_load(gen, __ATOMIC_ACQUIRE, __HIP_MEMORY_SCOPE_AGENT) == g) {
        __builtin_amdgcn_s_sleep(1);
      }
    }
  }
  __syncthreads();
  __threadfence();                 // acquire: invalidate stale cached h lines
}

// ---------------------------------------------------------------- persistent RNN
// 64 blocks: 4 M-groups (64 batch rows) x 16 N-groups (64 hid cols).
// Per block: 4 M-tiles x 4 N-tiles; wave w -> N-tile (w&3), M-tile pair (w>>2).
__global__ __launch_bounds__(NTHR)
void rnn_persistent(const float* __restrict__ x,
                    const float* __restrict__ Wout,
                    unsigned char* __restrict__ ws) {
  extern __shared__ _Float16 U[];   // [64 rows][ULD] combined activation slab

  unsigned* sync = (unsigned*)(ws + WS_SYNC);
  const float* bias = (const float*)(ws + WS_BIAS);
  float* z = (float*)(ws + WS_Z);
  _Float16* hA = (_Float16*)(ws + WS_H0);
  _Float16* hB = (_Float16*)(ws + WS_H1);
  const _Float16* Wc = (const _Float16*)(ws + WS_WC);

  const int tid   = threadIdx.x;
  const int lane  = tid & 31;
  const int wv    = tid >> 5;          // wave 0..7
  const int laneM = lane & 15;
  const bool hi   = lane >= 16;
  const int mt    = blockIdx.x & 3;    // M-group: 64 batch rows
  const int cg    = blockIdx.x >> 2;   // N-group 0..15: 64 hid cols
  const int mbase = mt * 64;
  const int ntile = wv & 3;            // N-tile within block
  const int mpair = wv >> 2;           // 0/1 -> M-tiles {2*mpair, 2*mpair+1}
  const int m0    = mpair * 2;
  const int m1    = m0 + 1;
  const int ncol  = cg * 64 + ntile * 16 + laneM;   // this lane's N column

  // zero the K-pad region once (never overwritten)
  for (int i = tid; i < 64 * 8; i += NTHR)
    U[(i >> 3) * ULD + (HID + INDIM) + (i & 7)] = (_Float16)0.f;

  const int asel = hi ? 8 : 0;        // A-fragment K phase per ISA layout
  const int bsel = hi ? 8 : 0;        // B-fragment K phase per ISA layout
  const float woutn = Wout[ncol];
  const float biasn = bias[ncol];
  const _Float16* wcp = Wc + (size_t)ncol * KP + bsel;
  const _Float16* ua0 = &U[(m0 * 16 + laneM) * ULD + asel];
  const _Float16* ua1 = &U[(m1 * 16 + laneM) * ULD + asel];

  for (int t = 0; t < T_SEQ; ++t) {
    const _Float16* hprev = (t & 1) ? hB : hA;
    _Float16*       hnext = (t & 1) ? hA : hB;
    const float* xt = x + (size_t)t * BATCH * INDIM;

    // ---- h region (64x1024 f16): async global -> LDS copy, 16B per lane.
    // ASYNCcnt-tracked; overlaps with the x-region conversion below.
    for (int i = tid; i < 64 * 128; i += NTHR) {
      int m = i >> 7, q = i & 127;
      unsigned lds_off = (unsigned)(size_t)&U[m * ULD + q * 8];
      unsigned long long ga =
          (unsigned long long)(size_t)&hprev[(size_t)(mbase + m) * HID + q * 8];
      asm volatile("global_load_async_to_lds_b128 %0, %1, off"
                   :: "v"(lds_off), "v"(ga) : "memory");
    }
    // ---- x region: convert f32 -> f16 (64x248) through VGPRs
    for (int i = tid; i < 64 * 62; i += NTHR) {
      int m = i / 62, q = i - m * 62;
      const float4 xv = *(const float4*)&xt[(size_t)(mbase + m) * INDIM + q * 4];
      v4h p = {(_Float16)xv.x, (_Float16)xv.y, (_Float16)xv.z, (_Float16)xv.w};
      *(v4h*)&U[m * ULD + HID + q * 4] = p;
    }
    asm volatile("s_wait_asynccnt 0x0" ::: "memory");
    __syncthreads();

    // ---- prefetch next step's x slab into cache during compute
    if (t + 1 < T_SEQ) {
      const float* xn = x + (size_t)(t + 1) * BATCH * INDIM
                          + (size_t)mbase * INDIM;
      if (tid < 248) __builtin_prefetch(xn + (size_t)tid * 64, 0, 1);
    }

    // ---- K loop: 40 chunks; B fragment loaded once, reused for 2 WMMAs
    v8f c0 = {};
    v8f c1 = {};
#pragma unroll 4
    for (int k0 = 0; k0 < KP; k0 += 32) {
      v8h bl  = *(const v8h*)(wcp + k0);
      v8h bh  = *(const v8h*)(wcp + k0 + 16);
      v8h a0l = *(const v8h*)(ua0 + k0);
      v8h a0h = *(const v8h*)(ua0 + k0 + 16);
      v8h a1l = *(const v8h*)(ua1 + k0);
      v8h a1h = *(const v8h*)(ua1 + k0 + 16);
      v16h B, A0, A1;
#pragma unroll
      for (int j = 0; j < 8; ++j) {
        B[j]  = bl[j];  B[j + 8]  = bh[j];
        A0[j] = a0l[j]; A0[j + 8] = a0h[j];
        A1[j] = a1l[j]; A1[j + 8] = a1h[j];
      }
      c0 = __builtin_amdgcn_wmma_f32_16x16x32_f16(false, A0, false, B,
                                                  (short)0, c0, false, false);
      c1 = __builtin_amdgcn_wmma_f32_16x16x32_f16(false, A1, false, B,
                                                  (short)0, c1, false, false);
    }

    // ---- epilogue: v_tanh, store h_t (f16), fused W_out reduction -> z[t]
#pragma unroll
    for (int r = 0; r < 8; ++r) {
      int rr = r + (hi ? 8 : 0);
      {
        int Mrow = mbase + m0 * 16 + rr;
        float v = TANH(c0[r] + biasn);
        hnext[(size_t)Mrow * HID + ncol] = (_Float16)v;
        float p = v * woutn;
        p += __shfl_xor(p, 1);
        p += __shfl_xor(p, 2);
        p += __shfl_xor(p, 4);
        p += __shfl_xor(p, 8);
        if (laneM == 0) atomicAdd(&z[t * BATCH + Mrow], p);
      }
      {
        int Mrow = mbase + m1 * 16 + rr;
        float v = TANH(c1[r] + biasn);
        hnext[(size_t)Mrow * HID + ncol] = (_Float16)v;
        float p = v * woutn;
        p += __shfl_xor(p, 1);
        p += __shfl_xor(p, 2);
        p += __shfl_xor(p, 4);
        p += __shfl_xor(p, 8);
        if (laneM == 0) atomicAdd(&z[t * BATCH + Mrow], p);
      }
    }

    grid_sync(&sync[0], &sync[1]);
  }
}

// ---------------------------------------------------------------- output scan
__global__ void out_scan(const unsigned char* __restrict__ ws,
                         const float* __restrict__ b_out,
                         const float* __restrict__ w_r,
                         const float* __restrict__ b_r,
                         float* __restrict__ out) {
  const float* z = (const float*)(ws + WS_Z);
  int b = threadIdx.x;
  float bo = b_out[0], wr = w_r[0], br = b_r[0];
  float oprev = 0.f;
  for (int t = 0; t < T_SEQ; ++t) {
    float zv = z[t * BATCH + b] + bo;
    float arg = (t == 0) ? zv : (zv + wr * oprev + br);
    float o = 1.f / (1.f + expf(-arg));
    out[b * T_SEQ + t] = o;
    oprev = o;
  }
}

// ---------------------------------------------------------------- launch
extern "C" void kernel_launch(void* const* d_in, const int* in_sizes, int n_in,
                              void* d_out, int out_size, void* d_ws, size_t ws_size,
                              hipStream_t stream) {
  const float* x     = (const float*)d_in[0];
  const float* W_ih  = (const float*)d_in[1];
  const float* b_ih  = (const float*)d_in[2];
  const float* W_hh  = (const float*)d_in[3];
  const float* b_hh  = (const float*)d_in[4];
  const float* W_out = (const float*)d_in[5];
  const float* b_out = (const float*)d_in[6];
  const float* w_r   = (const float*)d_in[7];
  const float* b_r   = (const float*)d_in[8];
  unsigned char* ws  = (unsigned char*)d_ws;
  float* out = (float*)d_out;

  hipLaunchKernelGGL(setup_wc, dim3((HID * KP + 255) / 256), dim3(256), 0, stream,
                     W_hh, W_ih, ws);
  hipLaunchKernelGGL(setup_misc, dim3((T_SEQ * BATCH + 255) / 256), dim3(256), 0, stream,
                     b_ih, b_hh, ws);
  hipLaunchKernelGGL(rnn_persistent, dim3(NBLK), dim3(NTHR), LDS_BYTES, stream,
                     x, W_out, ws);
  hipLaunchKernelGGL(out_scan, dim3(1), dim3(BATCH), 0, stream,
                     ws, b_out, w_r, b_r, out);
}
